// Model_89850715832971
// MI455X (gfx1250) — compile-verified, matrix operand order checked by reading, not run on hardware
//
#include <hip/hip_runtime.h>
#include <hip/hip_bf16.h>

typedef __attribute__((ext_vector_type(16))) _Float16 v16h;
typedef __attribute__((ext_vector_type(8)))  float    v8f;
typedef __attribute__((ext_vector_type(4)))  int      v4i;

static_assert(sizeof(v16h) == 32, "v16h must be 32 bytes");

// ---- CDNA5 async global->LDS path (guarded; falls back to sync staging) ----
#if __has_builtin(__builtin_amdgcn_global_load_async_to_lds_b128)
#define HAVE_ASYNC_LDS 1
#else
#define HAVE_ASYNC_LDS 0
#endif

// builtin signature (from hipcc diagnostic): (v4i __device__*, v4i __shared__*, Ii, Ii)
static __device__ __forceinline__ v4i* glb_v4(const void* p) {
  return (v4i*)p;
}
static __device__ __forceinline__ v4i* lds_v4(void* p) {
  return (v4i*)p;
}

static __device__ __forceinline__ void wait_async_lds() {
#if HAVE_ASYNC_LDS
#if __has_builtin(__builtin_amdgcn_s_wait_asynccnt)
  __builtin_amdgcn_s_wait_asynccnt(0);
#else
  asm volatile("s_wait_asynccnt 0x0" ::: "memory");
#endif
#endif
}

// -------------------- elementwise helpers --------------------

__global__ void f32_to_f16_kernel(const float* __restrict__ in,
                                  _Float16* __restrict__ out, int n) {
  int i = blockIdx.x * blockDim.x + threadIdx.x;
  if (i < n) out[i] = (_Float16)in[i];
}

// x (b,c,t,h,w) f32  ->  xh[f=(b*16+t)][c][n=h*w] f16
__global__ void x_to_frames_kernel(const float* __restrict__ x,
                                   _Float16* __restrict__ xh) {
  int o = blockIdx.x * 256 + threadIdx.x;       // 0 .. 16777215
  int pos = o & 1023;
  int ch  = (o >> 10) & 511;
  int f   = o >> 19;
  int b = f >> 4, tt = f & 15;
  long src = (((long)(b * 512 + ch) * 16 + tt) << 10) + pos;
  xh[o] = (_Float16)x[src];
}

// softmax over d (64) for q rows of qkv[f][0..511][1024]; *0.125 scale
__global__ void softmax_q_kernel(const _Float16* __restrict__ qkv,
                                 _Float16* __restrict__ qsm) {
  int gid = blockIdx.x * 256 + threadIdx.x;     // 262144 = f*8192 + head*1024 + n
  int n = gid & 1023;
  int head = (gid >> 10) & 7;
  int f = gid >> 13;
  const _Float16* base = qkv + (long)f * 1572864 + head * 65536 + n;
  float v[64];
  float mx = -1e30f;
#pragma unroll
  for (int d = 0; d < 64; ++d) {
    v[d] = (float)base[d * 1024];
    mx = fmaxf(mx, v[d]);
  }
  float s = 0.f;
#pragma unroll
  for (int d = 0; d < 64; ++d) { v[d] = expf(v[d] - mx); s += v[d]; }
  float inv = 0.125f / s;
  _Float16* ob = qsm + (long)f * 524288 + head * 65536 + n;
#pragma unroll
  for (int d = 0; d < 64; ++d) ob[d * 1024] = (_Float16)(v[d] * inv);
}

// softmax over n (1024) for k rows (qkv rows 512..1023); one wave per row
__global__ void softmax_k_kernel(const _Float16* __restrict__ qkv,
                                 _Float16* __restrict__ ksm) {
  int wr = blockIdx.x * 8 + (threadIdx.x >> 5);  // 16384 rows
  int lane = threadIdx.x & 31;
  int f = wr >> 9, r = wr & 511;
  const _Float16* base = qkv + (long)f * 1572864 + (512 + r) * 1024;
  float v[32];
  float mx = -1e30f;
#pragma unroll
  for (int j = 0; j < 32; ++j) {
    v[j] = (float)base[j * 32 + lane];
    mx = fmaxf(mx, v[j]);
  }
#pragma unroll
  for (int m = 16; m >= 1; m >>= 1) mx = fmaxf(mx, __shfl_xor(mx, m, 32));
  float s = 0.f;
#pragma unroll
  for (int j = 0; j < 32; ++j) { v[j] = expf(v[j] - mx); s += v[j]; }
#pragma unroll
  for (int m = 16; m >= 1; m >>= 1) s += __shfl_xor(s, m, 32);
  float inv = 1.f / s;
  _Float16* ob = ksm + (long)f * 524288 + r * 1024;
#pragma unroll
  for (int j = 0; j < 32; ++j) ob[j * 32 + lane] = (_Float16)(v[j] * inv);
}

// temporal attention core: per (b,s,head,i), t=16, d=64, softmax over j
// qkv layout: [row=(b*1024+s)*16+t][1536] (q|k|v, each head*64+d)
__global__ __launch_bounds__(256)
void temporal_attn_kernel(const _Float16* __restrict__ qkv,
                          _Float16* __restrict__ outa) {
  int gid = blockIdx.x * 256 + threadIdx.x;  // 262144 = bs*128 + head*16 + i
  int i = gid & 15;
  int head = (gid >> 4) & 7;
  int bs = gid >> 7;                          // 0..2047
  long rowbase = (long)bs * 16 * 1536;
  const _Float16* q = qkv + rowbase + (long)i * 1536 + head * 64;
  float qr[64];
#pragma unroll
  for (int d = 0; d < 64; ++d) qr[d] = (float)q[d];
  float sims[16];
  float mx = -1e30f;
#pragma unroll
  for (int j = 0; j < 16; ++j) {
    const _Float16* kp = qkv + rowbase + (long)j * 1536 + 512 + head * 64;
    float s = 0.f;
#pragma unroll
    for (int d = 0; d < 64; ++d) s += qr[d] * (float)kp[d];
    s *= 0.125f;
    sims[j] = s;
    mx = fmaxf(mx, s);
  }
  float denom = 0.f;
#pragma unroll
  for (int j = 0; j < 16; ++j) { sims[j] = expf(sims[j] - mx); denom += sims[j]; }
  float o[64];
#pragma unroll
  for (int d = 0; d < 64; ++d) o[d] = 0.f;
#pragma unroll
  for (int j = 0; j < 16; ++j) {
    const _Float16* vp = qkv + rowbase + (long)j * 1536 + 1024 + head * 64;
    float w = sims[j];
#pragma unroll
    for (int d = 0; d < 64; ++d) o[d] += w * (float)vp[d];
  }
  float inv = 1.f / denom;
  _Float16* ob = outa + (long)(bs * 16 + i) * 512 + head * 64;
#pragma unroll
  for (int d = 0; d < 64; ++d) ob[d] = (_Float16)(o[d] * inv);
}

// -------------------- generic batched WMMA GEMM --------------------
// C[MxN] = A[MxK] * B[KxN]; A row-major f16; B row-major (BT=false) or
// stored transposed as Bt[NxK] row-major (BT=true).  f32 accumulate.
// Tile 128x128, 8 waves (4x2), each wave 32x64 = 2x4 wmma 16x16 tiles.
// A-tile and Bt-tile staging uses GLOBAL_LOAD_ASYNC_TO_LDS_B128 when
// available (ASYNCcnt-tracked, bypasses VGPRs); falls back to sync staging.

enum { ST_F16 = 0, ST_F32 = 1, ST_H1 = 2, ST_RES = 3 };

template <bool BT, int SMODE>
__global__ __launch_bounds__(256)
void gemm_wmma_kernel(const _Float16* __restrict__ A,
                      const _Float16* __restrict__ B,
                      void* __restrict__ Cout,
                      const float* __restrict__ bias,
                      const float* __restrict__ xres,
                      int M, int N, int K, int lda, int ldb, int ldc,
                      long saO, long saI, long sbO, long sbI,
                      long scO, long scI, int zdiv) {
  __shared__ _Float16 ldsA[2][128][32];   // [m][k]
  __shared__ _Float16 ldsB[2][128][32];   // [n][k]

  const int z = blockIdx.z;
  const int zo = z / zdiv, zi = z - zo * zdiv;
  A += (long)zo * saO + (long)zi * saI;
  B += (long)zo * sbO + (long)zi * sbI;
  const long cbase = (long)zo * scO + (long)zi * scI;

  const int tid = threadIdx.x;
  const int lane = tid & 31;
  const int wid = tid >> 5;
  const int wm = (wid >> 1) * 32;
  const int wn = (wid & 1) * 64;
  const int tileM = blockIdx.y * 128;
  const int tileN = blockIdx.x * 128;

  v8f acc[2][4];
#pragma unroll
  for (int rt = 0; rt < 2; ++rt)
#pragma unroll
    for (int ct = 0; ct < 4; ++ct) acc[rt][ct] = (v8f){};

  auto load_tiles = [&](int buf, int k0) {
    // A tile: 4096 halves = 512 x b128 chunks, 2 per thread
#pragma unroll
    for (int i = 0; i < 2; ++i) {
      int c16 = tid * 2 + i;
      int off = c16 * 8;
      int r = off >> 5;
      int cc = off & 31;
      _Float16* dst = &ldsA[buf][r][cc];
#if HAVE_ASYNC_LDS
      if (tileM + r < M) {
        __builtin_amdgcn_global_load_async_to_lds_b128(
            glb_v4(A + (long)(tileM + r) * lda + k0 + cc), lds_v4(dst), 0, 0);
      } else {
        *(uint4*)dst = uint4{0u, 0u, 0u, 0u};
      }
#else
      uint4 val{0u, 0u, 0u, 0u};
      if (tileM + r < M)
        val = *(const uint4*)(A + (long)(tileM + r) * lda + k0 + cc);
      *(uint4*)dst = val;
#endif
    }
    if (BT) {
      // Bt is [N][K] row-major: direct b128 copy into [n][k]
#pragma unroll
      for (int i = 0; i < 2; ++i) {
        int c16 = tid * 2 + i;
        int off = c16 * 8;
        int r = off >> 5;
        int cc = off & 31;
        _Float16* dst = &ldsB[buf][r][cc];
#if HAVE_ASYNC_LDS
        if (tileN + r < N) {
          __builtin_amdgcn_global_load_async_to_lds_b128(
              glb_v4(B + (long)(tileN + r) * ldb + k0 + cc), lds_v4(dst), 0, 0);
        } else {
          *(uint4*)dst = uint4{0u, 0u, 0u, 0u};
        }
#else
        uint4 val{0u, 0u, 0u, 0u};
        if (tileN + r < N)
          val = *(const uint4*)(B + (long)(tileN + r) * ldb + k0 + cc);
        *(uint4*)dst = val;
#endif
      }
    } else {
      // B is [K][N]: read 8 halves along n, scatter into [n][k]
#pragma unroll
      for (int i = 0; i < 2; ++i) {
        int c16 = tid * 2 + i;
        int n0 = (c16 & 15) * 8;
        int kk = c16 >> 4;
        union { uint4 u; _Float16 h[8]; } t;
        t.u = uint4{0u, 0u, 0u, 0u};
        if (tileN + n0 < N)
          t.u = *(const uint4*)(B + (long)(k0 + kk) * ldb + tileN + n0);
#pragma unroll
        for (int j = 0; j < 8; ++j) ldsB[buf][n0 + j][kk] = t.h[j];
      }
    }
  };

  load_tiles(0, 0);
  wait_async_lds();
  __syncthreads();

  const int nk = K >> 5;
  for (int ks = 0; ks < nk; ++ks) {
    const int buf = ks & 1;
    if (ks + 1 < nk) load_tiles(buf ^ 1, (ks + 1) * 32);

    union Frag { v16h v; uint4 u[2]; };
    const int mrow = lane & 15;
    const int kb = (lane < 16) ? 0 : 8;

    Frag af[2];
#pragma unroll
    for (int rt = 0; rt < 2; ++rt) {
      const _Float16* p = &ldsA[buf][wm + rt * 16 + mrow][0];
      af[rt].u[0] = *(const uint4*)(p + kb);
      af[rt].u[1] = *(const uint4*)(p + kb + 16);
    }
    Frag bf[4];
#pragma unroll
    for (int ct = 0; ct < 4; ++ct) {
      const _Float16* p = &ldsB[buf][wn + ct * 16 + mrow][0];
      bf[ct].u[0] = *(const uint4*)(p + kb);
      bf[ct].u[1] = *(const uint4*)(p + kb + 16);
    }
#pragma unroll
    for (int rt = 0; rt < 2; ++rt)
#pragma unroll
      for (int ct = 0; ct < 4; ++ct)
        acc[rt][ct] = __builtin_amdgcn_wmma_f32_16x16x32_f16(
            false, af[rt].v, false, bf[ct].v, (short)0, acc[rt][ct], false,
            false);
    wait_async_lds();
    __syncthreads();
  }

  // ---- store (with fused epilogues) ----
  const int rowBase = tileM + wm + ((lane >> 4) << 3);
  const int colBase = tileN + wn + (lane & 15);
#pragma unroll
  for (int rt = 0; rt < 2; ++rt)
#pragma unroll
    for (int ct = 0; ct < 4; ++ct)
#pragma unroll
      for (int vr = 0; vr < 8; ++vr) {
        int m = rowBase + rt * 16 + vr;
        int n = colBase + ct * 16;
        if (m >= M || n >= N) continue;
        float val = acc[rt][ct][vr];
        if constexpr (SMODE == ST_F16) {
          ((_Float16*)Cout)[cbase + (long)m * ldc + n] = (_Float16)val;
        } else if constexpr (SMODE == ST_F32) {
          ((float*)Cout)[cbase + (long)m * ldc + n] = val;
        } else if constexpr (SMODE == ST_H1) {
          // spatial out-proj: +bias, transpose into xt[b][s=n][t][c=m] f16
          int f = zo, b = f >> 4, tt = f & 15;
          long idx = (((long)(b * 1024 + n) * 16 + tt) << 9) + m;
          ((_Float16*)Cout)[idx] = (_Float16)(val + bias[m]);
        } else {  // ST_RES: temporal out-proj + residual into d_out (f32)
          int b = m >> 14, s = (m >> 4) & 1023, tt = m & 15;
          long idx = (((long)(b * 512 + n) * 16 + tt) << 10) + s;
          ((float*)Cout)[idx] = val + xres[idx];
        }
      }
}

// -------------------- workspace layout (f16 elements) --------------------
#define OFF_WSQKV 0L
#define OFF_WSOUT 786432L
#define OFF_WTQKV 1048576L
#define OFF_WTOUT 1835008L
#define OFF_XH    2097152L      // 32*512*1024 ; reused as temporal attn output
#define OFF_QKV   18874368L     // 32*1536*1024 ; reused as temporal qkv
#define OFF_QSM   69206016L
#define OFF_KSM   85983232L
#define OFF_CTX   102760448L    // 32*8*64*64
#define OFF_OUTS  103809024L
#define OFF_XT    120586240L    // total end = 137363456 halves (~275 MB)

extern "C" void kernel_launch(void* const* d_in, const int* in_sizes, int n_in,
                              void* d_out, int out_size, void* d_ws,
                              size_t ws_size, hipStream_t stream) {
  (void)in_sizes; (void)n_in; (void)out_size; (void)ws_size;
  const float* x      = (const float*)d_in[0];
  const float* w_sqkv = (const float*)d_in[1];
  const float* w_sout = (const float*)d_in[2];
  const float* b_sout = (const float*)d_in[3];
  const float* w_tqkv = (const float*)d_in[4];
  const float* w_tout = (const float*)d_in[5];

  _Float16* ws = (_Float16*)d_ws;
  _Float16* wsqkv_h = ws + OFF_WSQKV;
  _Float16* wsout_h = ws + OFF_WSOUT;
  _Float16* wtqkv_h = ws + OFF_WTQKV;
  _Float16* wtout_h = ws + OFF_WTOUT;
  _Float16* xh   = ws + OFF_XH;     // spatial frames, later temporal attn out
  _Float16* qkvb = ws + OFF_QKV;    // spatial qkv, later temporal qkv
  _Float16* qsm  = ws + OFF_QSM;
  _Float16* ksm  = ws + OFF_KSM;
  _Float16* ctx  = ws + OFF_CTX;
  _Float16* outs = ws + OFF_OUTS;
  _Float16* xt   = ws + OFF_XT;

  dim3 blk(256);

  // weight + input conversion to f16
  f32_to_f16_kernel<<<dim3(3072), blk, 0, stream>>>(w_sqkv, wsqkv_h, 786432);
  f32_to_f16_kernel<<<dim3(1024), blk, 0, stream>>>(w_sout, wsout_h, 262144);
  f32_to_f16_kernel<<<dim3(3072), blk, 0, stream>>>(w_tqkv, wtqkv_h, 786432);
  f32_to_f16_kernel<<<dim3(1024), blk, 0, stream>>>(w_tout, wtout_h, 262144);
  x_to_frames_kernel<<<dim3(65536), blk, 0, stream>>>(x, xh);

  // spatial qkv: [1536x512] x [512x1024] per frame (32)
  gemm_wmma_kernel<false, ST_F16><<<dim3(8, 12, 32), blk, 0, stream>>>(
      wsqkv_h, xh, qkvb, nullptr, nullptr, 1536, 1024, 512, 512, 1024, 1024,
      0, 0, 524288, 0, 1572864, 0, 1);

  softmax_q_kernel<<<dim3(1024), blk, 0, stream>>>(qkvb, qsm);
  softmax_k_kernel<<<dim3(2048), blk, 0, stream>>>(qkvb, ksm);

  // ctx2[e][d] = sum_n v[e,n] * k[d,n]  (A = v rows, Bt = ksm), per (f,head)
  gemm_wmma_kernel<true, ST_F16><<<dim3(1, 1, 256), blk, 0, stream>>>(
      qkvb + 1048576, ksm, ctx, nullptr, nullptr, 64, 64, 1024, 1024, 1024, 64,
      1572864, 65536, 524288, 65536, 32768, 4096, 8);

  // out_attn = ctx2[64x64] x q[64x1024], per (f,head)
  gemm_wmma_kernel<false, ST_F16><<<dim3(8, 1, 256), blk, 0, stream>>>(
      ctx, qsm, outs, nullptr, nullptr, 64, 1024, 64, 64, 1024, 1024,
      32768, 4096, 524288, 65536, 524288, 65536, 8);

  // spatial out-proj + bias, fused transpose -> xt[b][s][t][c] f16
  gemm_wmma_kernel<false, ST_H1><<<dim3(8, 4, 32), blk, 0, stream>>>(
      wsout_h, outs, xt, b_sout, nullptr, 512, 1024, 512, 512, 1024, 0,
      0, 0, 524288, 0, 0, 0, 1);

  // temporal qkv: [32768x512] x [512x1536] (Bt = w_tqkv [1536x512])
  gemm_wmma_kernel<true, ST_F16><<<dim3(12, 256, 1), blk, 0, stream>>>(
      xt, wtqkv_h, qkvb, nullptr, nullptr, 32768, 1536, 512, 512, 512, 1536,
      0, 0, 0, 0, 0, 0, 1);

  // temporal softmax attention (t=16 per (b,s,head)) -> attn rows [32768x512]
  temporal_attn_kernel<<<dim3(1024), blk, 0, stream>>>(qkvb, xh);

  // temporal out-proj (Bt = w_tout) + residual, transposed store into d_out
  gemm_wmma_kernel<true, ST_RES><<<dim3(4, 256, 1), blk, 0, stream>>>(
      xh, wtout_h, d_out, nullptr, x, 32768, 512, 512, 512, 512, 0,
      0, 0, 0, 0, 0, 0, 1);
}
